// PointDiffusionTransformer_146028888618
// MI455X (gfx1250) — matrix operand order repaired and checked by reading, not compiled, not run on hardware
//
#include <hip/hip_runtime.h>
#include <hip/hip_bf16.h>

// ---------------------------------------------------------------------------
// PointDiffusionTransformer forward for MI455X (gfx1250, wave32, WMMA + TDM).
//
// Roofline: ~2.9 TFLOP total vs well under 1 GB of HBM traffic @ 23.3 TB/s ->
// compute bound. Use v_wmma_f32_16x16x32_bf16 (K=32 per op vs K=4 for the
// f32 WMMA) with fp32 accumulation; softmax/LN/GELU/residual in fp32.
// Cross-lane softmax reductions use DPP16 butterflies (pure VALU, no LDS).
// bf16 tile staging (GEMM A tile, attention K block) goes through the Tensor
// Data Mover (tensor_load_to_lds + s_wait_tensorcnt) with D# LDS padding.
//
// Workspace layout (144 MB):
//   [0,16MB)   a_bf   : LN output, bf16 [8192 x 1024]
//   [16,64MB)  qkv_bf : QKV GEMM output, bf16 [8192 x 3072]
//   [64,80MB)  at_bf  : attention output, bf16 [8192 x 1024]
//   [80,144MB) fc_bf  : GELU(FC) output, bf16 [8192 x 4096]
// The running residual h lives in d_out (fp32 [8192 x 1024]).
// ---------------------------------------------------------------------------

typedef __attribute__((ext_vector_type(16))) __bf16 v16bf;
typedef __attribute__((ext_vector_type(8)))  __bf16 v8bf;
typedef __attribute__((ext_vector_type(4)))  __bf16 v4bf;
typedef __attribute__((ext_vector_type(8)))  float  v8f;
typedef __attribute__((ext_vector_type(4)))  float  v4f;
typedef __attribute__((ext_vector_type(4)))  unsigned int v4u;
typedef __attribute__((ext_vector_type(8)))  int    v8i;
typedef __attribute__((ext_vector_type(4)))  int    v4i;

union ABf { v16bf v; v8bf h[2]; };

#if __has_builtin(__builtin_amdgcn_tensor_load_to_lds)
#define USE_TDM 1
#else
#define USE_TDM 0
#endif

#define W_DIM 1024
#define ROWS  8192   // B*N
#define BM 128
#define BN 128
#define BK 32
#define LDA 48       // As stride in elements (32 + 16 pad, keeps 16B alignment)
#define LDB 48       // Bs stride in elements

// ---------------- DPP16 cross-lane reductions (16-lane groups) -------------
// ctrl: 0xB1 = quad_perm(1,0,3,2)=xor1, 0x4E = quad_perm(2,3,0,1)=xor2,
//       0x141 = row_half_mirror (quad swap in 8), 0x140 = row_mirror (8-swap).
template<int CTRL>
__device__ __forceinline__ float dppf(float x) {
    const int xi = __builtin_bit_cast(int, x);
    const int r  = __builtin_amdgcn_update_dpp(xi, xi, CTRL, 0xF, 0xF, false);
    return __builtin_bit_cast(float, r);
}
__device__ __forceinline__ float red16_max(float x) {
    x = fmaxf(x, dppf<0x0B1>(x));
    x = fmaxf(x, dppf<0x04E>(x));
    x = fmaxf(x, dppf<0x141>(x));
    x = fmaxf(x, dppf<0x140>(x));
    return x;
}
__device__ __forceinline__ float red16_sum(float x) {
    x += dppf<0x0B1>(x);
    x += dppf<0x04E>(x);
    x += dppf<0x141>(x);
    x += dppf<0x140>(x);
    return x;
}

// --------------------------- TDM descriptor helper -------------------------
// 2-D bf16 tile load: tile_d0 elements/row, tile_d1 rows, global row stride
// 'gstride' (elements), optional LDS row padding via pad_interval/pad_amount.
// pi: pad_interval code (row bytes = 8 << pi must hold), pa: pad DWORDs - 1.
#if USE_TDM
__device__ __forceinline__ void tdm_load_2d_bf16(
    const __bf16* gptr, unsigned lds_off, int tile_d0, int tile_d1,
    int gstride, int pi, int pa, bool pad)
{
    v4u g0;
    v8i g1;
    const unsigned long long ga = (unsigned long long)(size_t)gptr;
    g0[0] = 1u;                                            // count=1, user D#
    g0[1] = lds_off;                                       // lds_addr
    g0[2] = (unsigned)ga;                                  // global_addr lo
    g0[3] = (unsigned)((ga >> 32) & 0x01FFFFFFu) | (2u << 30); // hi | type=2
    g1[0] = (1 << 16)                                      // data_size = 2B
          | ((pad ? 1 : 0) << 20)                          // pad_enable
          | (pi << 22) | (pa << 25);                       // pad interval/amt
    g1[1] = (int)((unsigned)tile_d0 << 16);                // tensor_dim0 lo16
    g1[2] = (int)((unsigned)tile_d1 << 16);                // tensor_dim1 lo16
    g1[3] = (int)((unsigned)tile_d0 << 16);                // tile_dim0
    g1[4] = tile_d1;                                       // tile_dim1
    g1[5] = gstride;                                       // dim0 stride lo32
    g1[6] = 0;
    g1[7] = 0;
    const v4i zg4 = (v4i){0, 0, 0, 0};
    const v8i zg8 = (v8i){0, 0, 0, 0, 0, 0, 0, 0};
    __builtin_amdgcn_tensor_load_to_lds(g0, g1, zg4, zg4, zg8, 0);
}
#endif

// ------------------------------- LayerNorm ---------------------------------
__global__ __launch_bounds__(256)
void ln_bf16_kernel(const float* __restrict__ x, const float* __restrict__ g,
                    const float* __restrict__ be, __bf16* __restrict__ out)
{
    __shared__ float red[256];
    const int row = blockIdx.x;
    const int t = threadIdx.x;
    const float* xr = x + (size_t)row * W_DIM;
    float v[4];
    float s = 0.f;
#pragma unroll
    for (int i = 0; i < 4; ++i) { v[i] = xr[t + 256 * i]; s += v[i]; }
    red[t] = s; __syncthreads();
    for (int o = 128; o > 0; o >>= 1) { if (t < o) red[t] += red[t + o]; __syncthreads(); }
    const float mu = red[0] * (1.0f / W_DIM);
    __syncthreads();
    float s2 = 0.f;
#pragma unroll
    for (int i = 0; i < 4; ++i) { float d = v[i] - mu; s2 += d * d; }
    red[t] = s2; __syncthreads();
    for (int o = 128; o > 0; o >>= 1) { if (t < o) red[t] += red[t + o]; __syncthreads(); }
    const float rs = rsqrtf(red[0] * (1.0f / W_DIM) + 1e-5f);
    __bf16* outr = out + (size_t)row * W_DIM;
#pragma unroll
    for (int i = 0; i < 4; ++i) {
        const int c = t + 256 * i;
        outr[c] = (__bf16)((v[i] - mu) * rs * g[c] + be[c]);
    }
}

// ------------------------------- GEMM (WMMA) -------------------------------
// C[M,N] = A[M,K](bf16) * B[K,N](fp32, staged->bf16) + bias, then epilogue.
// EPI: 0 = store bf16, 1 = exact GELU -> bf16, 2 = residual add -> fp32.
// A tile (bf16) is staged by the TDM; B tile is converted fp32->bf16 by VALU.
template<int EPI>
__global__ __launch_bounds__(256)
void gemm_bf16(const __bf16* __restrict__ A, const float* __restrict__ B,
               const float* __restrict__ bias, const float* __restrict__ resid,
               float* __restrict__ outF, __bf16* __restrict__ outB,
               int M, int N, int K)
{
    __shared__ __bf16 As[BM * LDA];
    __shared__ __bf16 Bs[BN * LDB];
    const int t = threadIdx.x;
    const int lane = t & 31, wave = t >> 5;
    const int wm = wave & 3, wn = wave >> 2;        // 4 waves in M, 2 in N
    const int rowBlk = blockIdx.y * BM;
    const int colBlk = blockIdx.x * BN;
    const int rsel = lane & 15;
    const int k0 = (lane >> 4) << 3;                // A frag K base: 0 or 8
    const int kr = (lane >> 4) << 4;                // unused: B frag also uses k0 (0 or 8)
    const int hf = lane >> 4;
#if USE_TDM
    const unsigned ldsAs = (unsigned)(size_t)(void*)As;
#endif

    v8f acc[2][4];
#pragma unroll
    for (int mi = 0; mi < 2; ++mi)
#pragma unroll
        for (int ni = 0; ni < 4; ++ni)
            acc[mi][ni] = (v8f){0.f,0.f,0.f,0.f,0.f,0.f,0.f,0.f};

    for (int kt = 0; kt < K; kt += BK) {
#if USE_TDM
        // A tile 128x32 bf16 via Tensor Data Mover; LDS rows padded 64B->96B
        // (pad_interval code 3 = 16 DWORDs = 64B row, pad 8 DWORDs = 32B).
        if (wave == 0)
            tdm_load_2d_bf16(A + (size_t)rowBlk * K + kt, ldsAs,
                             BK, BM, K, /*pi=*/3, /*pa=*/7, /*pad=*/true);
#else
        // stage A tile (bf16): 128 rows x 32, 2 x 16B per thread
#pragma unroll
        for (int i = 0; i < 2; ++i) {
            const int s = t + 256 * i;
            const int r = s >> 2, sg = s & 3;
            *(v8bf*)(As + r * LDA + sg * 8) =
                *(const v8bf*)(A + (size_t)(rowBlk + r) * K + kt + sg * 8);
        }
#endif
        // stage B tile (fp32 -> bf16, transposed to [n][k])
#pragma unroll
        for (int i = 0; i < 4; ++i) {
            const int vI = t + 256 * i;
            const int kk = vI >> 5, c4 = (vI & 31) * 4;
            v4f f = *(const v4f*)(B + (size_t)(kt + kk) * N + colBlk + c4);
            Bs[(c4 + 0) * LDB + kk] = (__bf16)f.x;
            Bs[(c4 + 1) * LDB + kk] = (__bf16)f.y;
            Bs[(c4 + 2) * LDB + kk] = (__bf16)f.z;
            Bs[(c4 + 3) * LDB + kk] = (__bf16)f.w;
        }
        if (kt + BK < K)   // next weight tile -> global_prefetch (stays in L2)
            __builtin_prefetch(B + (size_t)(kt + BK) * N + colBlk + (t & 31) * 4, 0, 1);
#if USE_TDM
        if (wave == 0)
            __builtin_amdgcn_s_wait_tensorcnt(0);
#endif
        __syncthreads();

        ABf af[2], bfr[4];
#pragma unroll
        for (int mi = 0; mi < 2; ++mi) {
            const __bf16* p = As + (wm * 32 + mi * 16 + rsel) * LDA + k0;
            af[mi].h[0] = *(const v8bf*)p;          // K = k0..k0+7
            af[mi].h[1] = *(const v8bf*)(p + 16);   // K = k0+16..k0+23
        }
#pragma unroll
        for (int ni = 0; ni < 4; ++ni) {
            const __bf16* p = Bs + (wn * 64 + ni * 16 + rsel) * LDB + k0;
            bfr[ni].h[0] = *(const v8bf*)p;         // K = k0..k0+7
            bfr[ni].h[1] = *(const v8bf*)(p + 16);  // K = k0+16..k0+23
        }
#pragma unroll
        for (int mi = 0; mi < 2; ++mi)
#pragma unroll
            for (int ni = 0; ni < 4; ++ni)
                acc[mi][ni] = __builtin_amdgcn_wmma_f32_16x16x32_bf16(
                    false, af[mi].v, false, bfr[ni].v, (short)0, acc[mi][ni],
                    false, false);
        __syncthreads();
    }

    // epilogue (C layout: VGPR v -> row v + 8*hf, lane%16 -> col)
#pragma unroll
    for (int mi = 0; mi < 2; ++mi)
#pragma unroll
        for (int ni = 0; ni < 4; ++ni) {
            const int colG = colBlk + wn * 64 + ni * 16 + rsel;
            const float bv = bias[colG];
#pragma unroll
            for (int v = 0; v < 8; ++v) {
                const int rowG = rowBlk + wm * 32 + mi * 16 + v + 8 * hf;
                float x = acc[mi][ni][v] + bv;
                const size_t idx = (size_t)rowG * N + colG;
                if (EPI == 0) {
                    outB[idx] = (__bf16)x;
                } else if (EPI == 1) {
                    x = 0.5f * x * (1.0f + erff(x * 0.70710678118654752f));
                    outB[idx] = (__bf16)x;
                } else {
                    outF[idx] = resid[idx] + x;
                }
            }
        }
}

// --------------------------- Flash attention -------------------------------
// qkv bf16 [B,N,3W] with head h at channels [h*192, h*192+192): q|k|v of 64.
// Grid: (N/64, H, B); 4 waves, each owns 16 query rows. Online softmax with
// one combined rescale per 32-key block; DPP16 row reductions. K block is
// staged by the TDM; V block is transposed manually (TDM cannot transpose).
__global__ __launch_bounds__(128)
void attn_kernel(const __bf16* __restrict__ qkv, __bf16* __restrict__ outB)
{
    const int qb = blockIdx.x, h = blockIdx.y, b = blockIdx.z;
    const int t = threadIdx.x, lane = t & 31, wave = t >> 5;
    const int rsel = lane & 15;
    const int k0 = (lane >> 4) << 3;
    const int kr = (lane >> 4) << 4;
    const int hf = lane >> 4;

    __shared__ __bf16 Ks[32 * 72];       // [key][c], stride 72 (144B, 16B-aligned)
    __shared__ __bf16 Vts[64 * 40];      // [c][key], stride 40 (80B, 16B-aligned)
    __shared__ __bf16 Pb[4 * 16 * 32];   // per-wave P tile [row][key]

    const size_t baseBN = (size_t)b * 1024 * 3072;
#if USE_TDM
    const unsigned ldsKs = (unsigned)(size_t)(void*)Ks;
#endif

    // Q fragments (2 K-chunks of 32), loaded straight from global in A layout
    ABf qf[2];
    {
        const int qrow = qb * 64 + wave * 16 + rsel;
        const __bf16* qp = qkv + baseBN + (size_t)qrow * 3072 + h * 192;
        qf[0].h[0] = *(const v8bf*)(qp + k0);
        qf[0].h[1] = *(const v8bf*)(qp + k0 + 16);
        qf[1].h[0] = *(const v8bf*)(qp + 32 + k0);
        qf[1].h[1] = *(const v8bf*)(qp + 32 + k0 + 16);
    }

    float m_run[8], l_run[8];
    v8f oacc[4];
#pragma unroll
    for (int v = 0; v < 8; ++v) { m_run[v] = -1e30f; l_run[v] = 0.f; }
#pragma unroll
    for (int ct = 0; ct < 4; ++ct)
        oacc[ct] = (v8f){0.f,0.f,0.f,0.f,0.f,0.f,0.f,0.f};

    for (int kb = 0; kb < 1024; kb += 32) {
#if USE_TDM
        // K block 32x64 bf16 via TDM; LDS rows padded 128B->144B
        // (pad_interval code 4 = 32 DWORDs = 128B row, pad 4 DWORDs = 16B).
        if (wave == 0)
            tdm_load_2d_bf16(qkv + baseBN + (size_t)kb * 3072 + h * 192 + 64,
                             ldsKs, 64, 32, 3072, /*pi=*/4, /*pa=*/3, true);
#else
        // stage K block [32 keys x 64c]
        {
            const int key = t >> 2, sg = t & 3;
            const __bf16* kp = qkv + baseBN + (size_t)(kb + key) * 3072
                             + h * 192 + 64 + sg * 16;
            *(v8bf*)(Ks + key * 72 + sg * 16)     = *(const v8bf*)kp;
            *(v8bf*)(Ks + key * 72 + sg * 16 + 8) = *(const v8bf*)(kp + 8);
        }
#endif
        // stage V block transposed -> [c][key]
#pragma unroll
        for (int i = 0; i < 4; ++i) {
            const int c = t + 128 * i;
            const int key = c >> 4, c0 = (c & 15) * 4;
            v4bf vv = *(const v4bf*)(qkv + baseBN + (size_t)(kb + key) * 3072
                                     + h * 192 + 128 + c0);
            Vts[(c0 + 0) * 40 + key] = vv.x;
            Vts[(c0 + 1) * 40 + key] = vv.y;
            Vts[(c0 + 2) * 40 + key] = vv.z;
            Vts[(c0 + 3) * 40 + key] = vv.w;
        }
#if USE_TDM
        if (wave == 0)
            __builtin_amdgcn_s_wait_tensorcnt(0);
#endif
        __syncthreads();

        // S = (Q*K^T) * (1/sqrt(64)); two 16-key subtiles computed up front
        v8f st[2];
#pragma unroll
        for (int nt = 0; nt < 2; ++nt) {
            v8f s = (v8f){0.f,0.f,0.f,0.f,0.f,0.f,0.f,0.f};
            ABf kf0, kf1;
            const __bf16* kpp = Ks + (nt * 16 + rsel) * 72 + k0;
            kf0.h[0] = *(const v8bf*)kpp;
            kf0.h[1] = *(const v8bf*)(kpp + 16);
            kf1.h[0] = *(const v8bf*)(kpp + 32);
            kf1.h[1] = *(const v8bf*)(kpp + 48);
            s = __builtin_amdgcn_wmma_f32_16x16x32_bf16(false, qf[0].v, false,
                    kf0.v, (short)0, s, false, false);
            s = __builtin_amdgcn_wmma_f32_16x16x32_bf16(false, qf[1].v, false,
                    kf1.v, (short)0, s, false, false);
            st[nt] = s;
        }

        // one online-softmax update per 32-key block (pure VALU reductions)
#pragma unroll
        for (int v = 0; v < 8; ++v) {
            const float s0 = st[0][v] * 0.125f;
            const float s1 = st[1][v] * 0.125f;
            const float mx = red16_max(fmaxf(s0, s1));
            const float mnew = fmaxf(m_run[v], mx);
            const float p0 = __expf(s0 - mnew);
            const float p1 = __expf(s1 - mnew);
            const float alpha = __expf(m_run[v] - mnew);
            const float sum = red16_sum(p0 + p1);
            l_run[v] = l_run[v] * alpha + sum;
            m_run[v] = mnew;
#pragma unroll
            for (int ct = 0; ct < 4; ++ct) oacc[ct][v] *= alpha;
            __bf16* pb = Pb + wave * 512 + (v + 8 * hf) * 32 + rsel;
            pb[0]  = (__bf16)p0;
            pb[16] = (__bf16)p1;
        }
        asm volatile("s_wait_dscnt 0" ::: "memory");   // wave-local LDS RAW

        ABf pf;
        const __bf16* pp = Pb + wave * 512 + rsel * 32 + k0;
        pf.h[0] = *(const v8bf*)pp;
        pf.h[1] = *(const v8bf*)(pp + 16);
#pragma unroll
        for (int ct = 0; ct < 4; ++ct) {
            ABf vf;
            const __bf16* vp = Vts + (ct * 16 + rsel) * 40 + k0;
            vf.h[0] = *(const v8bf*)vp;
            vf.h[1] = *(const v8bf*)(vp + 16);
            oacc[ct] = __builtin_amdgcn_wmma_f32_16x16x32_bf16(false, pf.v,
                          false, vf.v, (short)0, oacc[ct], false, false);
        }
        __syncthreads();
    }

#pragma unroll
    for (int ct = 0; ct < 4; ++ct)
#pragma unroll
        for (int v = 0; v < 8; ++v) {
            const int row = qb * 64 + wave * 16 + v + 8 * hf;
            const float val = oacc[ct][v] * (1.0f / l_run[v]);
            outB[(size_t)(b * 1024 + row) * 1024 + h * 64 + ct * 16 + rsel] =
                (__bf16)val;
        }
}

// ------------------------------- launcher ----------------------------------
extern "C" void kernel_launch(void* const* d_in, const int* in_sizes, int n_in,
                              void* d_out, int out_size, void* d_ws, size_t ws_size,
                              hipStream_t stream)
{
    const float* x     = (const float*)d_in[0];
    const float* ln1_g = (const float*)d_in[1];
    const float* ln1_b = (const float*)d_in[2];
    const float* w_qkv = (const float*)d_in[3];
    const float* b_qkv = (const float*)d_in[4];
    const float* w_o   = (const float*)d_in[5];
    const float* b_o   = (const float*)d_in[6];
    const float* ln2_g = (const float*)d_in[7];
    const float* ln2_b = (const float*)d_in[8];
    const float* w_fc  = (const float*)d_in[9];
    const float* b_fc  = (const float*)d_in[10];
    const float* w_pr  = (const float*)d_in[11];
    const float* b_pr  = (const float*)d_in[12];

    float* hbuf = (float*)d_out;                 // running residual, fp32
    char* ws = (char*)d_ws;
    __bf16* a_bf   = (__bf16*)(ws);                          // 16 MB
    __bf16* qkv_bf = (__bf16*)(ws + ((size_t)16 << 20));     // 48 MB
    __bf16* at_bf  = (__bf16*)(ws + ((size_t)64 << 20));     // 16 MB
    __bf16* fc_bf  = (__bf16*)(ws + ((size_t)80 << 20));     // 64 MB

    (void)hipMemcpyAsync(hbuf, x, (size_t)ROWS * W_DIM * sizeof(float),
                         hipMemcpyDeviceToDevice, stream);

    for (int l = 0; l < 12; ++l) {
        const float* g1  = ln1_g + (size_t)l * 1024;
        const float* bb1 = ln1_b + (size_t)l * 1024;
        const float* g2  = ln2_g + (size_t)l * 1024;
        const float* bb2 = ln2_b + (size_t)l * 1024;
        const float* wq  = w_qkv + (size_t)l * 1024 * 3072;
        const float* bq  = b_qkv + (size_t)l * 3072;
        const float* wo  = w_o   + (size_t)l * 1024 * 1024;
        const float* bo  = b_o   + (size_t)l * 1024;
        const float* wf  = w_fc  + (size_t)l * 1024 * 4096;
        const float* bf  = b_fc  + (size_t)l * 4096;
        const float* wp  = w_pr  + (size_t)l * 4096 * 1024;
        const float* bp  = b_pr  + (size_t)l * 1024;

        ln_bf16_kernel<<<ROWS, 256, 0, stream>>>(hbuf, g1, bb1, a_bf);

        gemm_bf16<0><<<dim3(3072 / BN, ROWS / BM), 256, 0, stream>>>(
            a_bf, wq, bq, nullptr, nullptr, qkv_bf, ROWS, 3072, 1024);

        attn_kernel<<<dim3(16, 16, 8), 128, 0, stream>>>(qkv_bf, at_bf);

        gemm_bf16<2><<<dim3(1024 / BN, ROWS / BM), 256, 0, stream>>>(
            at_bf, wo, bo, hbuf, hbuf, nullptr, ROWS, 1024, 1024);

        ln_bf16_kernel<<<ROWS, 256, 0, stream>>>(hbuf, g2, bb2, a_bf);

        gemm_bf16<1><<<dim3(4096 / BN, ROWS / BM), 256, 0, stream>>>(
            a_bf, wf, bf, nullptr, nullptr, fc_bf, ROWS, 4096, 1024);

        gemm_bf16<2><<<dim3(1024 / BN, ROWS / BM), 256, 0, stream>>>(
            fc_bf, wp, bp, hbuf, hbuf, nullptr, ROWS, 1024, 4096);
    }
}
